// ScaleDotProductAttention_266287972473
// MI455X (gfx1250) — compile-verified, matrix operand order checked by reading, not run on hardware
//
#include <hip/hip_runtime.h>
#include <hip/hip_bf16.h>

#define BB 16
#define HH 8
#define NN 1024
#define DD 64

typedef _Float16 v16h __attribute__((ext_vector_type(16)));
typedef _Float16 v8h  __attribute__((ext_vector_type(8)));
typedef float    v8f  __attribute__((ext_vector_type(8)));
typedef float    v4f  __attribute__((ext_vector_type(4)));

// Build a 16-half A/B fragment slice for V_WMMA_F32_16X16X32_F16 from fp32 memory.
// Fragment k-index for half h is: k = h + 8*(h>=8) (+ lane-half offset applied by
// caller), i.e. two contiguous 8-float runs at p[0..7] and p[16..23].
static __device__ __forceinline__ v16h pack_a16(const float* __restrict__ p) {
  v16h r;
#pragma unroll
  for (int i = 0; i < 8; ++i) r[i] = (_Float16)p[i];
#pragma unroll
  for (int i = 0; i < 8; ++i) r[8 + i] = (_Float16)p[16 + i];
  return r;
}

// Same fragment slice but from f16 memory: two aligned 16-byte loads + concat.
static __device__ __forceinline__ v16h pack_h16(const _Float16* __restrict__ p) {
  v8h a = *(const v8h*)(p);
  v8h b = *(const v8h*)(p + 16);
  return __builtin_shufflevector(a, b, 0, 1, 2, 3, 4, 5, 6, 7,
                                 8, 9, 10, 11, 12, 13, 14, 15);
}

// Max across the 16 lanes of a row: fused v_max_num_f32 with the DPP modifier
// on src0 -- one VALU instruction per butterfly step (no mov_dpp+canonicalize).
static __device__ __forceinline__ float rowmax16_dpp(float x) {
  float a = x, b;
  asm("v_max_num_f32_dpp %0, %1, %2 quad_perm:[1,0,3,2] row_mask:0xf bank_mask:0xf bound_ctrl:1"
      : "=v"(b) : "v"(a), "v"(a));
  asm("v_max_num_f32_dpp %0, %1, %2 quad_perm:[2,3,0,1] row_mask:0xf bank_mask:0xf bound_ctrl:1"
      : "=v"(a) : "v"(b), "v"(b));
  asm("v_max_num_f32_dpp %0, %1, %2 row_half_mirror row_mask:0xf bank_mask:0xf bound_ctrl:1"
      : "=v"(b) : "v"(a), "v"(a));
  asm("v_max_num_f32_dpp %0, %1, %2 row_mirror row_mask:0xf bank_mask:0xf bound_ctrl:1"
      : "=v"(a) : "v"(b), "v"(b));
  return a;
}

// ---------------------------------------------------------------------------
// Pre-pass 1: K -> f16 row-major [bh][n][d]; V -> f16 transposed [bh][d][n]
// ---------------------------------------------------------------------------
__global__ __launch_bounds__(256)
void cvt_kv_f16(const float* __restrict__ k, const float* __restrict__ v,
                _Float16* __restrict__ kf, _Float16* __restrict__ vt) {
  size_t idx = (size_t)blockIdx.x * 256 + threadIdx.x;  // over B*H*N*D
  float kv = k[idx];
  float vv = v[idx];
  kf[idx] = (_Float16)kv;  // coalesced
  size_t d  = idx % DD;
  size_t n  = (idx / DD) % NN;
  size_t bh = idx / ((size_t)DD * NN);
  vt[(bh * DD + d) * NN + n] = (_Float16)vv;  // transpose scatter (L2 absorbs)
}

// ---------------------------------------------------------------------------
// Pre-pass 2: pack 0/1 fp32 adjacency into bitmask words (32 cols -> 1 u32).
// ---------------------------------------------------------------------------
__global__ __launch_bounds__(256)
void pack_adj_bits(const float* __restrict__ adj, unsigned* __restrict__ adjm) {
  size_t idx = (size_t)blockIdx.x * 256 + threadIdx.x;  // over B*N*N
  float av = adj[idx];
  unsigned w = (unsigned)__ballot(av != 0.f);
  if ((threadIdx.x & 31) == 0) adjm[idx >> 5] = w;
}

// ---------------------------------------------------------------------------
// Main flash kernel, KT=64 key tiles. f16 K + f16 V^T + adj bitmask from ws.
// One wave = one 16-row Q tile for one head; 8 waves/block = 8 heads.
// V tiles staged into LDS with async-to-LDS copies, double buffered.
// ---------------------------------------------------------------------------
__global__ __launch_bounds__(256, 1)
void sdpa_flash_f16ws(const _Float16* __restrict__ kf,
                      const _Float16* __restrict__ vt,
                      const unsigned* __restrict__ adjm,
                      const float* __restrict__ q,
                      float* __restrict__ out) {
  // per wave (halves): V^T double buffer 2*4096 + P tile 1024 = 9216
  __shared__ __align__(16) _Float16 smem[8 * 9216];

  const int wave   = threadIdx.x >> 5;  // = head
  const int lane   = threadIdx.x & 31;
  const int lane15 = lane & 15;
  const int hlf    = lane >> 4;
  const int koff   = hlf * 8;   // fragment K offset for this half-wave
  const int rowoff = hlf * 8;   // C-fragment M offset for this half-wave

  const int b     = blockIdx.x / (NN / 16);
  const int qt    = blockIdx.x % (NN / 16);
  const int qbase = qt * 16;

  _Float16* wbase = smem + wave * 9216;  // vbuf[2] then ptile
  _Float16* ptile = wbase + 8192;
  const uint32_t lds_wbase = (uint32_t)(uintptr_t)wbase;  // LDS byte offset

  const size_t bh = (size_t)b * HH + wave;
  const float*    qp     = q  + (bh * NN + qbase + lane15) * DD;
  const _Float16* kbh    = kf + bh * (size_t)NN * DD;
  const _Float16* vtbh   = vt + bh * (size_t)DD * NN;
  const unsigned* adjm_b = adjm + (size_t)b * NN * (NN / 32);

  // Async copy of transposed V tile (64 d x 64 keys f16 = 8KB): 16 wave-instrs.
  const unsigned long long vt_bytes = (unsigned long long)(uintptr_t)vtbh;
  auto issueV = [&](int kbn, int bufsel) {
    uint32_t lb = lds_wbase + (uint32_t)bufsel * 8192u;
    unsigned long long gb = vt_bytes + (unsigned long long)kbn * 2ull;
#pragma unroll
    for (int i = 0; i < 16; ++i) {
      int chunk = i * 32 + lane;  // 512 x 16B chunks
      int d = chunk >> 3, sub = chunk & 7;
      unsigned long long ga = gb + (unsigned long long)d * (NN * 2)
                                 + (unsigned long long)(sub * 16);
      uint32_t la = lb + (uint32_t)chunk * 16u;
      asm volatile("global_load_async_to_lds_b128 %0, %1, off"
                   :: "v"(la), "v"(ga) : "memory");
    }
  };

  // Q A-fragments (16x64 -> two 16x32 f16 fragments), converted once
  v16h aq[2];
#pragma unroll
  for (int c = 0; c < 2; ++c) aq[c] = pack_a16(qp + c * 32 + koff);

  // Constant B fragment whose column 0 is all-ones: P @ [V | 1] row-sum trick.
  v16h bvone;
#pragma unroll
  for (int i = 0; i < 16; ++i) bvone[i] = (lane15 == 0) ? (_Float16)1 : (_Float16)0;

  // o[0..3] = output d-chunks, o[4] column 0 = softmax denominator
  v8f o[5];
  float m[8], md[8];  // true running max (-inf aware) and sanitized copy
#pragma unroll
  for (int dc = 0; dc < 5; ++dc)
#pragma unroll
    for (int r = 0; r < 8; ++r) o[dc][r] = 0.f;
#pragma unroll
  for (int r = 0; r < 8; ++r) { m[r] = -INFINITY; md[r] = 0.f; }

  issueV(0, 0);  // prime the pipeline

  for (int kb = 0; kb < NN; kb += 64) {
    const int buf = (kb >> 6) & 1;
    int kbn = kb + 64;
    if (kbn >= NN) kbn = NN - 64;  // clamp: harmless re-fetch on last iter
    issueV(kbn, buf ^ 1);
    __builtin_prefetch(kbh + (size_t)(kbn + lane) * DD, 0, 3);  // next K tile

    // ---- adjacency: one aligned u32 pair covers this row's 64-key tile
    unsigned mwlo[8], mwhi[8];
    {
      const unsigned* aw = adjm_b + (size_t)(qbase + rowoff) * (NN / 32) + (kb >> 5);
#pragma unroll
      for (int r = 0; r < 8; ++r) {
        uint2 w = *(const uint2*)(aw + (size_t)r * (NN / 32));
        mwlo[r] = w.x;
        mwhi[r] = w.y;
      }
    }

    // ---- S = Q @ K^T : four 16x16 col-groups, K from f16 ws (b128 loads)
    v8f s[4];
#pragma unroll
    for (int g = 0; g < 4; ++g) {
#pragma unroll
      for (int i = 0; i < 8; ++i) s[g][i] = 0.f;
      const _Float16* krow = kbh + (size_t)(kb + g * 16 + lane15) * DD;
#pragma unroll
      for (int c = 0; c < 2; ++c) {
        v16h bk = pack_h16(krow + c * 32 + koff);
        s[g] = __builtin_amdgcn_wmma_f32_16x16x32_f16(
            false, aq[c], false, bk, (short)0, s[g], false, false);
      }
    }

    // ---- mask + tile row-max (fused DPP 16-lane reduction, no LDS)
    float tmax[8];
#pragma unroll
    for (int r = 0; r < 8; ++r) tmax[r] = -INFINITY;
#pragma unroll
    for (int g = 0; g < 4; ++g)
#pragma unroll
      for (int r = 0; r < 8; ++r) {
        unsigned w = (g < 2) ? mwlo[r] : mwhi[r];
        bool on = (w >> ((g & 1) * 16 + lane15)) & 1u;
        float sv = on ? s[g][r] : -INFINITY;
        s[g][r] = sv;
        tmax[r] = fmaxf(tmax[r], sv);
      }
#pragma unroll
    for (int r = 0; r < 8; ++r) tmax[r] = rowmax16_dpp(tmax[r]);

    // ---- online softmax rescale via sanitized max (no per-element guards)
    float scale[8];
#pragma unroll
    for (int r = 0; r < 8; ++r) {
      float mn  = fmaxf(m[r], tmax[r]);
      float mdn = (mn == -INFINITY) ? 0.f : mn;
      scale[r] = __expf(fminf(md[r] - mdn, 0.f));  // dead rows: o==0, any finite ok
      m[r] = mn;
      md[r] = mdn;
    }
    // p = exp(s - md): masked lanes are -inf -> exp gives exact 0
#pragma unroll
    for (int g = 0; g < 4; ++g)
#pragma unroll
      for (int r = 0; r < 8; ++r) s[g][r] = __expf(s[g][r] - md[r]);
#pragma unroll
    for (int dc = 0; dc < 5; ++dc)
#pragma unroll
      for (int r = 0; r < 8; ++r) o[dc][r] *= scale[r];

    // ---- P (C layout) -> LDS -> reload as A fragments (16x64 f16)
#pragma unroll
    for (int g = 0; g < 4; ++g)
#pragma unroll
      for (int r = 0; r < 8; ++r)
        ptile[(r + rowoff) * 64 + g * 16 + lane15] = (_Float16)s[g][r];

    asm volatile("s_wait_dscnt 0" ::: "memory");        // P visible in-wave
    asm volatile("s_wait_asynccnt 0x10" ::: "memory");  // tile kb resident in LDS

    v16h ap[2];
#pragma unroll
    for (int kc = 0; kc < 2; ++kc)
      ap[kc] = pack_h16(ptile + lane15 * 64 + kc * 32 + koff);

    // ---- O += P @ V_tile from transposed LDS tile: 2 x b128 per (dc,kc)
    const _Float16* vb = wbase + buf * 4096;
#pragma unroll
    for (int dc = 0; dc < 4; ++dc)
#pragma unroll
      for (int kc = 0; kc < 2; ++kc) {
        v16h bv = pack_h16(vb + (dc * 16 + lane15) * 64 + kc * 32 + koff);
        o[dc] = __builtin_amdgcn_wmma_f32_16x16x32_f16(
            false, ap[kc], false, bv, (short)0, o[dc], false, false);
      }
#pragma unroll
    for (int kc = 0; kc < 2; ++kc)
      o[4] = __builtin_amdgcn_wmma_f32_16x16x32_f16(
          false, ap[kc], false, bvone, (short)0, o[4], false, false);
  }

  // ---- epilogue: denominators live in column 0 of o[4] (lane 0 / lane 16)
  float invl[8];
#pragma unroll
  for (int r = 0; r < 8; ++r) {
    float rs = __shfl(o[4][r], hlf * 16, 32);
    invl[r] = (rs > 0.f) ? (1.f / rs) : 0.f;  // fully-masked rows -> 0
  }
  float* op = out + (bh * NN + qbase + rowoff) * DD;
#pragma unroll
  for (int dc = 0; dc < 4; ++dc)
#pragma unroll
    for (int r = 0; r < 8; ++r)
      op[(size_t)r * DD + dc * 16 + lane15] = o[dc][r] * invl[r];
}

// ---------------------------------------------------------------------------
// Fallback (no ws needed): fp32 loads + in-loop conversion.
// ---------------------------------------------------------------------------
__global__ __launch_bounds__(256, 1)
void sdpa_flash_fallback(const float* __restrict__ q,
                         const float* __restrict__ k,
                         const float* __restrict__ v,
                         const float* __restrict__ adj,
                         float* __restrict__ out) {
  __shared__ __align__(16) _Float16 smem[8 * (32 * 64 + 16 * 32)];
  const int wave = threadIdx.x >> 5;
  const int lane = threadIdx.x & 31;
  const int lane15 = lane & 15;
  const int hlf = lane >> 4;
  const int koff = hlf * 8, rowoff = hlf * 8;
  const int b = blockIdx.x / (NN / 16);
  const int qt = blockIdx.x % (NN / 16);
  const int qbase = qt * 16;
  _Float16* vtile = smem + wave * (32 * 64 + 16 * 32);
  _Float16* ptile = vtile + 32 * 64;
  const size_t bh = (size_t)b * HH + wave;
  const float* qp = q + (bh * NN + qbase + lane15) * DD;
  const float* kbase_p = k + bh * NN * DD;
  const float* vbase_p = v + bh * NN * DD;
  const float* adj_b = adj + (size_t)b * NN * NN;
  v16h aq[2];
#pragma unroll
  for (int c = 0; c < 2; ++c) aq[c] = pack_a16(qp + c * 32 + koff);
  v8f o[4];
  float m[8], lsum[8];
#pragma unroll
  for (int dc = 0; dc < 4; ++dc)
#pragma unroll
    for (int r = 0; r < 8; ++r) o[dc][r] = 0.f;
#pragma unroll
  for (int r = 0; r < 8; ++r) { m[r] = -INFINITY; lsum[r] = 0.f; }
  for (int kb = 0; kb < NN; kb += 32) {
    {
      const float* vrow = vbase_p + (size_t)(kb + lane) * DD;
      _Float16* dst = vtile + lane * 64;
#pragma unroll
      for (int d = 0; d < 64; d += 4) {
        v4f f = *(const v4f*)(vrow + d);
        dst[d + 0] = (_Float16)f[0];
        dst[d + 1] = (_Float16)f[1];
        dst[d + 2] = (_Float16)f[2];
        dst[d + 3] = (_Float16)f[3];
      }
    }
    v8f s[2];
#pragma unroll
    for (int g = 0; g < 2; ++g) {
#pragma unroll
      for (int i = 0; i < 8; ++i) s[g][i] = 0.f;
      const float* krow = kbase_p + (size_t)(kb + g * 16 + lane15) * DD;
#pragma unroll
      for (int c = 0; c < 2; ++c) {
        v16h bk = pack_a16(krow + c * 32 + koff);
        s[g] = __builtin_amdgcn_wmma_f32_16x16x32_f16(
            false, aq[c], false, bk, (short)0, s[g], false, false);
      }
    }
    float tmax[8];
#pragma unroll
    for (int r = 0; r < 8; ++r) tmax[r] = -INFINITY;
#pragma unroll
    for (int g = 0; g < 2; ++g) {
      const float* arow = adj_b + (size_t)(qbase + rowoff) * NN + kb + g * 16 + lane15;
#pragma unroll
      for (int r = 0; r < 8; ++r) {
        float av = arow[(size_t)r * NN];
        float sv = (av != 0.f) ? s[g][r] : -INFINITY;
        s[g][r] = sv;
        tmax[r] = fmaxf(tmax[r], sv);
      }
    }
#pragma unroll
    for (int r = 0; r < 8; ++r) tmax[r] = rowmax16_dpp(tmax[r]);
    float scale[8];
#pragma unroll
    for (int r = 0; r < 8; ++r) {
      float mn = fmaxf(m[r], tmax[r]);
      scale[r] = (m[r] == -INFINITY) ? 0.f : __expf(m[r] - mn);
      m[r] = mn;
    }
    float p[2][8], rsum[8];
#pragma unroll
    for (int r = 0; r < 8; ++r) rsum[r] = 0.f;
#pragma unroll
    for (int g = 0; g < 2; ++g)
#pragma unroll
      for (int r = 0; r < 8; ++r) {
        float pv = (m[r] != -INFINITY) ? __expf(s[g][r] - m[r]) : 0.f;
        p[g][r] = pv;
        rsum[r] += pv;
      }
#pragma unroll
    for (int r = 0; r < 8; ++r) {
      float t = rsum[r];
      t += __shfl_xor(t, 1, 32);
      t += __shfl_xor(t, 2, 32);
      t += __shfl_xor(t, 4, 32);
      t += __shfl_xor(t, 8, 32);
      lsum[r] = lsum[r] * scale[r] + t;
    }
#pragma unroll
    for (int dc = 0; dc < 4; ++dc)
#pragma unroll
      for (int r = 0; r < 8; ++r) o[dc][r] *= scale[r];
#pragma unroll
    for (int g = 0; g < 2; ++g)
#pragma unroll
      for (int r = 0; r < 8; ++r)
        ptile[(r + rowoff) * 32 + g * 16 + lane15] = (_Float16)p[g][r];
    asm volatile("s_wait_dscnt 0" ::: "memory");
    v16h ap = pack_h16(ptile + lane15 * 32 + koff);
#pragma unroll
    for (int dc = 0; dc < 4; ++dc) {
      v16h bv;
#pragma unroll
      for (int i = 0; i < 16; ++i) {
        int kk = i + ((i >= 8) ? 8 : 0) + koff;
        bv[i] = vtile[kk * 64 + dc * 16 + lane15];
      }
      o[dc] = __builtin_amdgcn_wmma_f32_16x16x32_f16(
          false, ap, false, bv, (short)0, o[dc], false, false);
    }
  }
  float invl[8];
#pragma unroll
  for (int r = 0; r < 8; ++r) invl[r] = (lsum[r] > 0.f) ? (1.f / lsum[r]) : 0.f;
  float* op = out + (bh * NN + qbase + rowoff) * DD;
#pragma unroll
  for (int dc = 0; dc < 4; ++dc)
#pragma unroll
    for (int r = 0; r < 8; ++r)
      op[(size_t)r * DD + dc * 16 + lane15] = o[dc][r] * invl[r];
}

extern "C" void kernel_launch(void* const* d_in, const int* in_sizes, int n_in,
                              void* d_out, int out_size, void* d_ws, size_t ws_size,
                              hipStream_t stream) {
  (void)in_sizes; (void)n_in; (void)out_size;
  const float* q   = (const float*)d_in[0];
  const float* k   = (const float*)d_in[1];
  const float* v   = (const float*)d_in[2];
  const float* adj = (const float*)d_in[3];
  float* out = (float*)d_out;

  const size_t elems     = (size_t)BB * HH * NN * DD;          // 8388608
  const size_t kv_bytes  = 2 * elems * sizeof(_Float16);       // 32 MB
  const size_t adj_words = (size_t)BB * NN * (NN / 32);        // 524288
  const size_t need      = kv_bytes + adj_words * sizeof(unsigned);
  dim3 grid(BB * (NN / 16)), block(256);

  if (ws_size >= need) {
    _Float16* kf = (_Float16*)d_ws;
    _Float16* vt = kf + elems;
    unsigned* adjm = (unsigned*)((char*)d_ws + kv_bytes);
    cvt_kv_f16<<<dim3((unsigned)(elems / 256)), block, 0, stream>>>(k, v, kf, vt);
    pack_adj_bits<<<dim3((unsigned)((size_t)BB * NN * NN / 256)), block, 0, stream>>>(adj, adjm);
    sdpa_flash_f16ws<<<grid, block, 0, stream>>>(kf, vt, adjm, q, out);
  } else {
    sdpa_flash_fallback<<<grid, block, 0, stream>>>(q, k, v, adj, out);
  }
}
